// VectorAttention_15298673508394
// MI455X (gfx1250) — compile-verified, matrix operand order checked by reading, not run on hardware
//
#include <hip/hip_runtime.h>

typedef __attribute__((ext_vector_type(16))) _Float16 v16h;
typedef __attribute__((ext_vector_type(8)))  float    v8f;
typedef __attribute__((ext_vector_type(4)))  unsigned int v4u;
typedef __attribute__((ext_vector_type(8)))  int v8i;
typedef __attribute__((ext_vector_type(4)))  int v4i;

#define BB   4
#define CC   128
#define DD   64
#define NN   2048
#define MM   2048
#define HH   256
#define KNN  16
#define NKK  (NN * KNN)   // 32768

#define TILE_ROWS 64
#define TILE_COLS 64
#define KSTEP     32

// ---- capability gates (probe-confirmed on this toolchain) -----------------
#if __has_builtin(__builtin_amdgcn_tensor_load_to_lds)
#define VA_HAS_TDM 1
#else
#define VA_HAS_TDM 0
#endif
#if __has_include(<hip/amd_detail/amd_gfx1250_TDM.h>)
#define VA_TDM_6ARG 1
#endif
#if __has_builtin(__builtin_amdgcn_global_load_async_to_lds_b128)
#define VA_HAS_ASYNC 1
#else
#define VA_HAS_ASYNC 0
#endif
// ---------------------------------------------------------------------------

// Typed AS casts for the async-LDS builtins: b128 variant takes v4i pointers
// in AS(1) (global) and AS(3) (LDS). LDS offset = low 32 bits of flat address.
#define VA_AS1_V4I(p) ((__attribute__((address_space(1))) v4i*)(uintptr_t)(p))
#define VA_AS3_V4I(p) ((__attribute__((address_space(3))) v4i*)(unsigned int)(uintptr_t)(p))

union HFrag { v16h v; _Float16 e[16]; };

// Per-lane K index inside a 16x32 (A) / 32x16 (B) f16 WMMA fragment (CDNA5 layout).
__device__ __forceinline__ int va_kmap(int lane, int h) {
  return (h & 7) + ((h & 8) << 1) + ((lane & 16) >> 1);
}

__device__ __forceinline__ v8f va_wmma(const HFrag& a, const HFrag& x, v8f acc) {
  return __builtin_amdgcn_wmma_f32_16x16x32_f16(false, a.v, false, x.v,
                                                (short)0, acc, false, false);
}

#if VA_HAS_ASYNC
__device__ __forceinline__ void va_wait_async() {
#if __has_builtin(__builtin_amdgcn_s_wait_asynccnt)
  __builtin_amdgcn_s_wait_asynccnt(0);
#else
  asm volatile("s_wait_asynccnt 0" ::: "memory");
#endif
}
#endif

// Store one 16x16 f32 C/D tile (+bias, optional residual).
__device__ __forceinline__ void va_store_tile(float* __restrict__ Y,
                                              const float* __restrict__ bias,
                                              const float* __restrict__ resid,
                                              v8f acc, int b, int Dout, int Ncols,
                                              int dbase, size_t colbase, int lane) {
  const int rbase = (lane & 16) >> 1;
  const int row = lane & 15;
#pragma unroll
  for (int r = 0; r < 8; ++r) {
    const int d = dbase + r + rbase;
    const size_t o = ((size_t)b * Dout + d) * Ncols + colbase + row;
    float vv = acc[r] + bias[d];
    if (resid) vv += resid[o];
    Y[o] = vv;
  }
}

#if VA_HAS_TDM
// Issue a TDM 2D tile load: tileK (contiguous) x tileRows, f32 elements,
// row stride = rowStrideElems, into LDS at `laddr`.
__device__ __forceinline__ void va_tdm_load_2d(const void* gaddr, void* laddr,
                                               int tileK, int tileRows,
                                               unsigned long long rowStrideElems,
                                               unsigned long long tensorW,
                                               unsigned long long tensorH) {
  const unsigned long long ga = (unsigned long long)(uintptr_t)gaddr;
  const unsigned int lds = (unsigned int)(uintptr_t)laddr;  // low 32 bits = LDS offset
  v4u g0;
  g0.x = 1u;                               // count=1 (valid user descriptor)
  g0.y = lds;                              // lds_addr (bytes)
  g0.z = (unsigned int)ga;                 // global_addr[31:0]   (D# bits 95:64)
  g0.w = (unsigned int)(ga >> 32) | (2u << 30);  // addr[56:32] | type=2

  unsigned int w[8] = {0u, 0u, 0u, 0u, 0u, 0u, 0u, 0u};
  w[0] = 2u << 16;                                   // data_size=2 -> 4 bytes
  w[1] |= (unsigned int)((tensorW & 0xFFFFull) << 16);   // tensor_dim0[15:0]
  w[2] |= (unsigned int)((tensorW >> 16) & 0xFFFFull);   // tensor_dim0[31:16]
  w[2] |= (unsigned int)((tensorH & 0xFFFFull) << 16);   // tensor_dim1[15:0]
  w[3] |= (unsigned int)((tensorH >> 16) & 0xFFFFull);   // tensor_dim1[31:16]
  w[3] |= (unsigned int)tileK << 16;                     // tile_dim0 @127:112
  w[4] |= (unsigned int)tileRows & 0xFFFFu;              // tile_dim1 @143:128
  w[5] = (unsigned int)(rowStrideElems & 0xFFFFFFFFull); // tensor_dim0_stride[31:0]
  w[6] |= (unsigned int)((rowStrideElems >> 32) & 0xFFFFull);
  v8i g1;
  g1[0] = (int)w[0]; g1[1] = (int)w[1]; g1[2] = (int)w[2]; g1[3] = (int)w[3];
  g1[4] = (int)w[4]; g1[5] = (int)w[5]; g1[6] = (int)w[6]; g1[7] = (int)w[7];
  v4i z4 = {0, 0, 0, 0};
#if defined(VA_TDM_6ARG)
  v8i z8 = {0, 0, 0, 0, 0, 0, 0, 0};
  __builtin_amdgcn_tensor_load_to_lds(g0, g1, z4, z4, z8, 0);
#else
  __builtin_amdgcn_tensor_load_to_lds(g0, g1, z4, z4, 0);
#endif
}
#endif  // VA_HAS_TDM

// ---------------------------------------------------------------------------
// LDS-staged GEMM:  Y[b,d,col] = sum_c W[d,c] * X[b,c,col] + bias[d] (+resid)
// 128 threads = 4 waves; block tile 64 rows x 64 cols; each wave computes a
// 16-row strip with 4 accumulators (A fragment reused across 4 WMMAs).
// A K-tile staged via Tensor Data Mover (wave 0); B K-tile staged via
// async global->LDS b128 copies (ASYNCcnt) when available.
// ---------------------------------------------------------------------------
__global__ void __launch_bounds__(128)
va_proj_gemm(const float* __restrict__ W, const float* __restrict__ bias,
             const float* __restrict__ X, const float* __restrict__ resid,
             float* __restrict__ Y, int Cin, int Ncols, int Dout) {
  __shared__ __align__(16) float sA[TILE_ROWS * KSTEP];   // [row][k]
  __shared__ __align__(16) float sB[KSTEP][TILE_COLS];    // [k][col]
  const int tid  = threadIdx.x;
  const int lane = tid & 31;
  const int wave = tid >> 5;
  const int n0 = blockIdx.x * TILE_COLS;
  const int d0 = blockIdx.y * TILE_ROWS;
  const int b  = blockIdx.z;
  const int row = lane & 15;
  const float* Xb = X + (size_t)b * Cin * Ncols;

  v8f acc0 = {}, acc1 = {}, acc2 = {}, acc3 = {};

  for (int c0 = 0; c0 < Cin; c0 += KSTEP) {
    // --- stage A tile (64 rows x 32 k, f32) via TDM ---
#if VA_HAS_TDM
    if (wave == 0)
      va_tdm_load_2d(&W[(size_t)d0 * Cin + c0], sA, KSTEP, TILE_ROWS,
                     (unsigned long long)Cin, (unsigned long long)Cin,
                     (unsigned long long)Dout);
#elif VA_HAS_ASYNC
#pragma unroll
    for (int i = 0; i < (TILE_ROWS * KSTEP / 4); i += 128) {
      const int e = (i + tid) * 4;
      const int r = e / KSTEP, kk = e % KSTEP;
      __builtin_amdgcn_global_load_async_to_lds_b128(
          VA_AS1_V4I(&W[(size_t)(d0 + r) * Cin + c0 + kk]),
          VA_AS3_V4I(&sA[r * KSTEP + kk]), 0, 0);
    }
#else
#pragma unroll
    for (int i = 0; i < (TILE_ROWS * KSTEP / 4); i += 128) {
      const int e = (i + tid) * 4;
      const int r = e / KSTEP, kk = e % KSTEP;
      *(float4*)&sA[r * KSTEP + kk] =
          *(const float4*)&W[(size_t)(d0 + r) * Cin + c0 + kk];
    }
#endif
    // --- stage B tile (32 k x 64 cols, f32) via async global->LDS ---
#if VA_HAS_ASYNC
#pragma unroll
    for (int i = 0; i < (KSTEP * TILE_COLS / 4); i += 128) {
      const int e = (i + tid) * 4;
      const int kk = e / TILE_COLS, cc = e % TILE_COLS;
      __builtin_amdgcn_global_load_async_to_lds_b128(
          VA_AS1_V4I(&Xb[(size_t)(c0 + kk) * Ncols + n0 + cc]),
          VA_AS3_V4I(&sB[kk][cc]), 0, 0);
    }
#else
#pragma unroll
    for (int i = 0; i < (KSTEP * TILE_COLS / 4); i += 128) {
      const int e = (i + tid) * 4;
      const int kk = e / TILE_COLS, cc = e % TILE_COLS;
      *(float4*)&sB[kk][cc] =
          *(const float4*)&Xb[(size_t)(c0 + kk) * Ncols + n0 + cc];
    }
#endif
    if (c0 + KSTEP < Cin)
      __builtin_prefetch(&Xb[(size_t)(c0 + KSTEP) * Ncols + n0], 0, 1);
#if VA_HAS_ASYNC
    va_wait_async();
#endif
#if VA_HAS_TDM
    if (wave == 0) __builtin_amdgcn_s_wait_tensorcnt(0);
#endif
    __syncthreads();

    HFrag a, x0, x1, x2, x3;
#pragma unroll
    for (int h = 0; h < 16; ++h) {
      const int kk = va_kmap(lane, h);
      a.e[h]  = (_Float16)sA[(wave * 16 + row) * KSTEP + kk];
      x0.e[h] = (_Float16)sB[kk][row];
      x1.e[h] = (_Float16)sB[kk][16 + row];
      x2.e[h] = (_Float16)sB[kk][32 + row];
      x3.e[h] = (_Float16)sB[kk][48 + row];
    }
    acc0 = va_wmma(a, x0, acc0);
    acc1 = va_wmma(a, x1, acc1);
    acc2 = va_wmma(a, x2, acc2);
    acc3 = va_wmma(a, x3, acc3);
    __syncthreads();
  }
  const int dbase = d0 + wave * 16;
  va_store_tile(Y, bias, resid, acc0, b, Dout, Ncols, dbase, (size_t)n0 +  0, lane);
  va_store_tile(Y, bias, resid, acc1, b, Dout, Ncols, dbase, (size_t)n0 + 16, lane);
  va_store_tile(Y, bias, resid, acc2, b, Dout, Ncols, dbase, (size_t)n0 + 32, lane);
  va_store_tile(Y, bias, resid, acc3, b, Dout, Ncols, dbase, (size_t)n0 + 48, lane);
}

// ---------------------------------------------------------------------------
// Brute-force 16-NN; support points staged in LDS, register top-16.
// ---------------------------------------------------------------------------
__global__ void va_knn(const float* __restrict__ pq, const float* __restrict__ ps,
                       int* __restrict__ idx) {
  __shared__ float sx[MM], sy[MM], sz[MM];
  const int b = blockIdx.y;
  const float* psb = ps + (size_t)b * 3 * MM;
  for (int i = threadIdx.x; i < MM; i += blockDim.x) {
    sx[i] = psb[i];
    sy[i] = psb[MM + i];
    sz[i] = psb[2 * MM + i];
  }
  __syncthreads();

  const int n = blockIdx.x * blockDim.x + threadIdx.x;
  const float* pqb = pq + (size_t)b * 3 * NN;
  const float qx = pqb[n], qy = pqb[NN + n], qz = pqb[2 * NN + n];

  float bd[KNN]; int bi[KNN];
#pragma unroll
  for (int j = 0; j < KNN; ++j) { bd[j] = 1e30f; bi[j] = 0; }
  float worst = 1e30f; int worstj = 0;

  for (int m = 0; m < MM; ++m) {
    const float dx = qx - sx[m], dy = qy - sy[m], dz = qz - sz[m];
    const float d = dx * dx + dy * dy + dz * dz;
    if (d < worst) {
#pragma unroll
      for (int j = 0; j < KNN; ++j)
        if (j == worstj) { bd[j] = d; bi[j] = m; }
      worst = bd[0]; worstj = 0;
#pragma unroll
      for (int j = 1; j < KNN; ++j)
        if (bd[j] > worst) { worst = bd[j]; worstj = j; }
    }
  }
  int* o = idx + ((size_t)b * NN + n) * KNN;
#pragma unroll
  for (int j = 0; j < KNN; ++j) o[j] = bi[j];
}

// ---------------------------------------------------------------------------
// pe_pre[b,d,col] = Wp1[d,:] . (pq[:,n] - ps[:,idx]) + bp1[d]   (col = n*K+k)
// ---------------------------------------------------------------------------
__global__ void va_pos_pre(const float* __restrict__ pq, const float* __restrict__ ps,
                           const int* __restrict__ idx, const float* __restrict__ Wp1,
                           const float* __restrict__ bp1, float* __restrict__ pe_pre) {
  const int col = blockIdx.x * blockDim.x + threadIdx.x;
  const int d = blockIdx.y, b = blockIdx.z;
  const int n = col / KNN, j = col % KNN;
  const int m = idx[((size_t)b * NN + n) * KNN + j];
  const float* pqb = pq + (size_t)b * 3 * NN;
  const float* psb = ps + (size_t)b * 3 * MM;
  const float rx = pqb[n] - psb[m];
  const float ry = pqb[NN + n] - psb[MM + m];
  const float rz = pqb[2 * NN + n] - psb[2 * MM + m];
  pe_pre[((size_t)b * DD + d) * NKK + col] =
      Wp1[d * 3 + 0] * rx + Wp1[d * 3 + 1] * ry + Wp1[d * 3 + 2] * rz + bp1[d];
}

// ---------------------------------------------------------------------------
// Per-channel sum / sumsq over (B, cols) of X laid out (B, Ch, NKK).
// ---------------------------------------------------------------------------
__global__ void va_bn_stats(const float* __restrict__ X, float* __restrict__ sum,
                            float* __restrict__ sumsq, int Ch) {
  const int col = blockIdx.x * blockDim.x + threadIdx.x;
  const int c = blockIdx.y;
  float s1 = 0.f, s2 = 0.f;
  for (int b = 0; b < BB; ++b) {
    const float x = X[((size_t)b * Ch + c) * NKK + col];
    s1 += x; s2 += x * x;
  }
  __shared__ float r1[256], r2[256];
  const int t = threadIdx.x;
  r1[t] = s1; r2[t] = s2;
  __syncthreads();
  for (int off = 128; off > 0; off >>= 1) {
    if (t < off) { r1[t] += r1[t + off]; r2[t] += r2[t + off]; }
    __syncthreads();
  }
  if (t == 0) {
    atomicAdd(&sum[c], r1[0]);
    atomicAdd(&sumsq[c], r2[0]);
  }
}

__global__ void va_bn_finalize(const float* __restrict__ sum, const float* __restrict__ sumsq,
                               float* __restrict__ mean, float* __restrict__ rstd,
                               int Ch, float inv_cnt) {
  const int c = blockIdx.x * blockDim.x + threadIdx.x;
  if (c < Ch) {
    const float mu = sum[c] * inv_cnt;
    const float var = sumsq[c] * inv_cnt - mu * mu;
    mean[c] = mu;
    rstd[c] = rsqrtf(var + 1e-5f);
  }
}

// ---------------------------------------------------------------------------
// Y = W . relu(gamma*(X-mean)*rstd + beta) + bias; one wave per 16x64 tile,
// BN+ReLU fused into the B-fragment build, A fragment reused across 4 WMMAs.
// ---------------------------------------------------------------------------
__global__ void __launch_bounds__(32)
va_bnrelu_gemm(const float* __restrict__ W, const float* __restrict__ bias,
               const float* __restrict__ X, const float* __restrict__ gamma,
               const float* __restrict__ beta, const float* __restrict__ mean,
               const float* __restrict__ rstd, float* __restrict__ Y,
               int Cin, int Ncols, int Dout) {
  const int lane = threadIdx.x;
  const int n0 = blockIdx.x * 64;
  const int d0 = blockIdx.y * 16;
  const int b  = blockIdx.z;
  const int row = lane & 15;
  const float* Xb = X + (size_t)b * Cin * Ncols;
  v8f acc0 = {}, acc1 = {}, acc2 = {}, acc3 = {};
  for (int c0 = 0; c0 < Cin; c0 += 32) {
    if (c0 + 32 < Cin)
      __builtin_prefetch(&Xb[(size_t)(c0 + 32) * Ncols + n0 + row], 0, 1);
    HFrag a, x0, x1, x2, x3;
#pragma unroll
    for (int h = 0; h < 16; ++h) {
      const int c = c0 + va_kmap(lane, h);
      a.e[h] = (_Float16)W[(size_t)(d0 + row) * Cin + c];
      const float sc = gamma[c] * rstd[c];
      const float sh = beta[c] - mean[c] * sc;
      const float* xr = &Xb[(size_t)c * Ncols + n0 + row];
      x0.e[h] = (_Float16)fmaxf(xr[0]  * sc + sh, 0.f);
      x1.e[h] = (_Float16)fmaxf(xr[16] * sc + sh, 0.f);
      x2.e[h] = (_Float16)fmaxf(xr[32] * sc + sh, 0.f);
      x3.e[h] = (_Float16)fmaxf(xr[48] * sc + sh, 0.f);
    }
    acc0 = va_wmma(a, x0, acc0);
    acc1 = va_wmma(a, x1, acc1);
    acc2 = va_wmma(a, x2, acc2);
    acc3 = va_wmma(a, x3, acc3);
  }
  va_store_tile(Y, bias, nullptr, acc0, b, Dout, Ncols, d0, (size_t)n0 +  0, lane);
  va_store_tile(Y, bias, nullptr, acc1, b, Dout, Ncols, d0, (size_t)n0 + 16, lane);
  va_store_tile(Y, bias, nullptr, acc2, b, Dout, Ncols, d0, (size_t)n0 + 32, lane);
  va_store_tile(Y, bias, nullptr, acc3, b, Dout, Ncols, d0, (size_t)n0 + 48, lane);
}

// ---------------------------------------------------------------------------
// a1_pre = Wa1 . (q - group(k) + pe) + ba1; B-fragment composed on the fly
// (gather via idx) so qk_rel/key_g are never materialized. 4 WMMAs per step.
// ---------------------------------------------------------------------------
__global__ void __launch_bounds__(32)
va_attn_gemm(const float* __restrict__ W, const float* __restrict__ bias,
             const float* __restrict__ q, const float* __restrict__ kbuf,
             const float* __restrict__ pe, const int* __restrict__ idx,
             float* __restrict__ Y, int Cin, int Ncols, int Dout) {
  const int lane = threadIdx.x;
  const int n0 = blockIdx.x * 64;
  const int d0 = blockIdx.y * 16;
  const int b  = blockIdx.z;
  const int row = lane & 15;
  int cols[4], ns[4], ms[4];
#pragma unroll
  for (int s = 0; s < 4; ++s) {
    cols[s] = n0 + s * 16 + row;
    ns[s] = cols[s] / KNN;
    ms[s] = idx[((size_t)b * NN + ns[s]) * KNN + (cols[s] % KNN)];
  }
  v8f acc0 = {}, acc1 = {}, acc2 = {}, acc3 = {};
  for (int c0 = 0; c0 < Cin; c0 += 32) {
    HFrag a, x0, x1, x2, x3;
#pragma unroll
    for (int h = 0; h < 16; ++h) {
      const int c = c0 + va_kmap(lane, h);
      a.e[h] = (_Float16)W[(size_t)(d0 + row) * Cin + c];
      const float* qr = &q[((size_t)b * Cin + c) * NN];
      const float* kr = &kbuf[((size_t)b * Cin + c) * MM];
      const float* pr = &pe[((size_t)b * Cin + c) * Ncols];
      x0.e[h] = (_Float16)(qr[ns[0]] - kr[ms[0]] + pr[cols[0]]);
      x1.e[h] = (_Float16)(qr[ns[1]] - kr[ms[1]] + pr[cols[1]]);
      x2.e[h] = (_Float16)(qr[ns[2]] - kr[ms[2]] + pr[cols[2]]);
      x3.e[h] = (_Float16)(qr[ns[3]] - kr[ms[3]] + pr[cols[3]]);
    }
    acc0 = va_wmma(a, x0, acc0);
    acc1 = va_wmma(a, x1, acc1);
    acc2 = va_wmma(a, x2, acc2);
    acc3 = va_wmma(a, x3, acc3);
  }
  va_store_tile(Y, bias, nullptr, acc0, b, Dout, Ncols, d0, (size_t)n0 +  0, lane);
  va_store_tile(Y, bias, nullptr, acc1, b, Dout, Ncols, d0, (size_t)n0 + 16, lane);
  va_store_tile(Y, bias, nullptr, acc2, b, Dout, Ncols, d0, (size_t)n0 + 32, lane);
  va_store_tile(Y, bias, nullptr, acc3, b, Dout, Ncols, d0, (size_t)n0 + 48, lane);
}

// ---------------------------------------------------------------------------
// softmax over K + aggregate: agg[b,d,n] = sum_k softmax(a2)_k * (v_gathered + pe)
// ---------------------------------------------------------------------------
__global__ void va_softmax_agg(const float* __restrict__ a2, const float* __restrict__ vbuf,
                               const float* __restrict__ pe, const int* __restrict__ idx,
                               float* __restrict__ agg) {
  const size_t t = (size_t)blockIdx.x * blockDim.x + threadIdx.x;  // over B*D*N
  const int n = (int)(t % NN);
  const int d = (int)((t / NN) % DD);
  const int b = (int)(t / ((size_t)NN * DD));
  const size_t rowo = ((size_t)b * DD + d) * NKK + (size_t)n * KNN;
  const int* irow = idx + ((size_t)b * NN + n) * KNN;
  float mx = -1e30f;
#pragma unroll
  for (int j = 0; j < KNN; ++j) mx = fmaxf(mx, a2[rowo + j]);
  float s = 0.f, out = 0.f;
#pragma unroll
  for (int j = 0; j < KNN; ++j) {
    const float e = __expf(a2[rowo + j] - mx);
    const float val = vbuf[((size_t)b * DD + d) * MM + irow[j]] + pe[rowo + j];
    s += e;
    out += e * val;
  }
  agg[((size_t)b * DD + d) * NN + n] = out / s;
}

// ---------------------------------------------------------------------------

extern "C" void kernel_launch(void* const* d_in, const int* in_sizes, int n_in,
                              void* d_out, int out_size, void* d_ws, size_t ws_size,
                              hipStream_t stream) {
  (void)in_sizes; (void)n_in; (void)out_size; (void)ws_size;
  const float* pq  = (const float*)d_in[0];
  const float* fq  = (const float*)d_in[1];
  const float* ps  = (const float*)d_in[2];
  const float* fs  = (const float*)d_in[3];
  const float* Wq  = (const float*)d_in[4];  const float* bq  = (const float*)d_in[5];
  const float* Wk  = (const float*)d_in[6];  const float* bk  = (const float*)d_in[7];
  const float* Wv  = (const float*)d_in[8];  const float* bv  = (const float*)d_in[9];
  const float* Wp1 = (const float*)d_in[10]; const float* bp1 = (const float*)d_in[11];
  const float* gp  = (const float*)d_in[12]; const float* btp = (const float*)d_in[13];
  const float* Wp2 = (const float*)d_in[14]; const float* bp2 = (const float*)d_in[15];
  const float* Wa1 = (const float*)d_in[16]; const float* ba1 = (const float*)d_in[17];
  const float* ga  = (const float*)d_in[18]; const float* bta = (const float*)d_in[19];
  const float* Wa2 = (const float*)d_in[20]; const float* ba2 = (const float*)d_in[21];
  const float* We  = (const float*)d_in[22]; const float* be  = (const float*)d_in[23];
  float* out = (float*)d_out;

  float* ws = (float*)d_ws;
  size_t off = 0;
  float* q_b   = ws + off; off += (size_t)BB * DD * NN;
  float* k_b   = ws + off; off += (size_t)BB * DD * MM;
  float* v_b   = ws + off; off += (size_t)BB * DD * MM;
  float* agg   = ws + off; off += (size_t)BB * DD * NN;
  int*   idx   = (int*)(ws + off); off += (size_t)BB * NN * KNN;
  float* posSum  = ws + off; off += DD;   // posSum/posSq contiguous for one memset
  float* posSq   = ws + off; off += DD;
  float* posMean = ws + off; off += DD;
  float* posRstd = ws + off; off += DD;
  float* aSum    = ws + off; off += HH;   // aSum/aSq contiguous for one memset
  float* aSq     = ws + off; off += HH;
  float* aMean   = ws + off; off += HH;
  float* aRstd   = ws + off; off += HH;
  float* pe_pre  = ws + off; off += (size_t)BB * DD * NKK;
  float* pe      = ws + off; off += (size_t)BB * DD * NKK;
  float* a1      = ws + off; off += (size_t)BB * HH * NKK;
  float* a2      = ws + off; off += (size_t)BB * DD * NKK;

  (void)hipMemsetAsync(posSum, 0, sizeof(float) * 2 * DD, stream);
  (void)hipMemsetAsync(aSum,   0, sizeof(float) * 2 * HH, stream);

  const float inv_cnt = 1.0f / (float)((size_t)BB * NKK);

  // q/k/v projections (LDS-staged WMMA, TDM weight tiles, async B tiles)
  va_proj_gemm<<<dim3(NN / 64, DD / 64, BB), 128, 0, stream>>>(Wq, bq, fq, nullptr, q_b, CC, NN, DD);
  va_proj_gemm<<<dim3(MM / 64, DD / 64, BB), 128, 0, stream>>>(Wk, bk, fs, nullptr, k_b, CC, MM, DD);
  va_proj_gemm<<<dim3(MM / 64, DD / 64, BB), 128, 0, stream>>>(Wv, bv, fs, nullptr, v_b, CC, MM, DD);

  // 16-NN
  va_knn<<<dim3(NN / 256, BB), 256, 0, stream>>>(pq, ps, idx);

  // pos_mlp: conv(3->D), BN stats, BN+ReLU fused into conv(D->D) WMMA GEMM
  va_pos_pre<<<dim3(NKK / 256, DD, BB), 256, 0, stream>>>(pq, ps, idx, Wp1, bp1, pe_pre);
  va_bn_stats<<<dim3(NKK / 256, DD), 256, 0, stream>>>(pe_pre, posSum, posSq, DD);
  va_bn_finalize<<<1, 256, 0, stream>>>(posSum, posSq, posMean, posRstd, DD, inv_cnt);
  va_bnrelu_gemm<<<dim3(NKK / 64, DD / 16, BB), 32, 0, stream>>>(
      Wp2, bp2, pe_pre, gp, btp, posMean, posRstd, pe, DD, NKK, DD);

  // attn_mlp: fused (q - group(k) + pe) -> conv(D->H) WMMA GEMM
  va_attn_gemm<<<dim3(NKK / 64, HH / 16, BB), 32, 0, stream>>>(
      Wa1, ba1, q_b, k_b, pe, idx, a1, DD, NKK, HH);
  va_bn_stats<<<dim3(NKK / 256, HH), 256, 0, stream>>>(a1, aSum, aSq, HH);
  va_bn_finalize<<<1, 256, 0, stream>>>(aSum, aSq, aMean, aRstd, HH, inv_cnt);
  va_bnrelu_gemm<<<dim3(NKK / 64, DD / 16, BB), 32, 0, stream>>>(
      Wa2, ba2, a1, ga, bta, aMean, aRstd, a2, HH, NKK, DD);

  // softmax over K + weighted aggregate of (group(v) + pe)
  va_softmax_agg<<<dim3((BB * DD * NN) / 256), 256, 0, stream>>>(a2, v_b, pe, idx, agg);

  // final projection D->C with residual add of fq
  va_proj_gemm<<<dim3(NN / 64, CC / 64, BB), 128, 0, stream>>>(We, be, agg, fq, out, DD, NN, CC);
}
